// SlotAttention_71734543778322
// MI455X (gfx1250) — matrix operand order repaired and checked by reading, not compiled, not run on hardware
//
#include <hip/hip_runtime.h>
#include <hip/hip_bf16.h>
#include <math.h>

typedef _Float16 half_t;
typedef __attribute__((ext_vector_type(16))) _Float16 v16h;
typedef __attribute__((ext_vector_type(8)))  _Float16 v8h;
typedef __attribute__((ext_vector_type(8)))  float    v8f;

#define B_SZ   32
#define NKV    4096
#define DIN    256
#define NQ     8
#define DS     256
#define DMLP   1024
#define NHEAD  4
#define DH     64
#define N_IT   3
#define EPS_A  1e-8f
#define LN_EPS 1e-5f

__device__ __forceinline__ v8f wmma_f16(v16h a, v16h b, v8f c) {
  // v_wmma_f32_16x16x32_f16 : D = A(16x32 f16) x B(32x16 f16) + C(f32)
  return __builtin_amdgcn_wmma_f32_16x16x32_f16(false, a, false, b, (short)0, c,
                                                false, false);
}

// A fragment: lane L -> row m=L&15, K = g*16+{0..15}, g=L>>4
__device__ __forceinline__ v16h load_a_frag(const half_t* base, int stride, int lane) {
  int m = lane & 15, g = lane >> 4;
  const half_t* p = base + (size_t)m * stride + g * 16;
  v16h a;
#pragma unroll
  for (int e = 0; e < 8; ++e) { a[e] = p[e]; a[e + 8] = p[8 + e]; }
  return a;
}

// B fragment from B^T (row-major [N][K]): lane L -> col n=L&15, K = g*16+e
__device__ __forceinline__ v16h load_bt_frag(const half_t* base, int stride, int lane) {
  int n = lane & 15, g = lane >> 4;
  const half_t* p = base + (size_t)n * stride + g * 16;
  v16h b;
#pragma unroll
  for (int e = 0; e < 16; ++e) b[e] = p[e];
  return b;
}

// ---------------- k0: weight fp16 conversion, slot copy, pad zeroing ----------------
__global__ void k0_prep(const float* __restrict__ Wk, const float* __restrict__ Wv,
                        const float* __restrict__ slots_init,
                        half_t* __restrict__ wk_h, half_t* __restrict__ wv_h,
                        half_t* __restrict__ q_h, half_t* __restrict__ a_t,
                        float* __restrict__ slots_buf) {
  int i = blockIdx.x * blockDim.x + threadIdx.x;            // 65536 threads
  wk_h[i] = (half_t)Wk[i];
  wv_h[i] = (half_t)Wv[i];
  slots_buf[i] = slots_init[i];
  if (i < 512) q_h[B_SZ * NHEAD * NQ * DH + i] = (half_t)0.f;               // q pad rows
  if (i < NQ * NKV) a_t[(size_t)B_SZ * NHEAD * NQ * NKV + i] = (half_t)0.f; // a_t pad
}

// ---------------- k1: LN(inputs) + K/V projection via WMMA ----------------
__global__ void k1_ln_kv(const float* __restrict__ inputs,
                         const float* __restrict__ gvec, const float* __restrict__ bvec,
                         const half_t* __restrict__ wk_h, const half_t* __restrict__ wv_h,
                         half_t* __restrict__ k_h, half_t* __restrict__ v_h) {
  union SmemU {                       // xs is dead after LN -> reuse as output stage
    float  xs[32][264];
    half_t st[2][32][264];            // [mat][row][col] fp16 D tiles
  };
  __shared__ __align__(16) SmemU uA;
  __shared__ half_t xh[32][272];
  int tid = threadIdx.x;
  size_t rowBase = (size_t)blockIdx.x * 32;
  const float* src = inputs + rowBase * DIN;
  for (int idx = tid; idx < 32 * DIN; idx += 256)
    uA.xs[idx >> 8][idx & 255] = src[idx];
  __syncthreads();
  // LN stats: 8 threads per row
  {
    int r = tid >> 3, j = tid & 7;
    float s = 0.f, s2 = 0.f;
#pragma unroll
    for (int i = 0; i < 32; ++i) { float x = uA.xs[r][j + 8 * i]; s += x; s2 += x * x; }
    for (int m = 1; m < 8; m <<= 1) { s += __shfl_xor(s, m, 8); s2 += __shfl_xor(s2, m, 8); }
    float mean = s * (1.f / 256.f);
    float var  = s2 * (1.f / 256.f) - mean * mean;
    float rstd = rsqrtf(var + LN_EPS);
#pragma unroll
    for (int i = 0; i < 32; ++i) {
      int c = j + 8 * i;
      xh[r][c] = (half_t)((uA.xs[r][c] - mean) * rstd * gvec[c] + bvec[c]);
    }
  }
  __syncthreads();                    // xs dead from here
  // WMMA: 8 waves; wave = (mat, row-tile, col-half); wave id is uniform -> SGPRs
  int w = __builtin_amdgcn_readfirstlane(tid >> 5);
  int lane = tid & 31;
  int mat = w >> 2, sub = w & 3;
  int rt = sub >> 1, cth = sub & 1;
  v16h af[8];
#pragma unroll
  for (int kc = 0; kc < 8; ++kc)
    af[kc] = load_a_frag(&xh[rt * 16][kc * 32], 272, lane);
  const half_t* W = mat ? wv_h : wk_h;
  float scale = mat ? 1.0f : 0.125f;  // k *= dh^-0.5
  int n = lane & 15, h8 = lane >> 4;
#pragma unroll 1
  for (int t = 0; t < 8; ++t) {       // constant trip count, scalar loop control
    int ct = cth * 8 + t;
    v8f acc = {};
#pragma unroll
    for (int kc = 0; kc < 8; ++kc) {
      v16h bf = load_bt_frag(W + (size_t)(ct * 16) * DIN + kc * 32, DIN, lane);
      acc = wmma_f16(af[kc], bf, acc);
    }
#pragma unroll
    for (int rr = 0; rr < 8; ++rr)
      uA.st[mat][rt * 16 + h8 * 8 + rr][ct * 16 + n] = (half_t)(acc[rr] * scale);
  }
  __syncthreads();
  // coalesced b128 writeout: 2048 chunks of 8 halfs
  for (int chunk = tid; chunk < 2 * 32 * 32; chunk += 256) {
    int mat2 = chunk >> 10;
    int r    = (chunk >> 5) & 31;
    int c8   = chunk & 31;
    half_t* dst = (mat2 ? v_h : k_h) + (rowBase + r) * (size_t)DS + c8 * 8;
    *(v8h*)dst = *(const v8h*)&uA.st[mat2][r][c8 * 8];
  }
}

// ---------------- k2: slot LN + q projection (fp32 VALU, tiny) ----------------
__global__ void k2_q(const float* __restrict__ slots_buf,
                     const float* __restrict__ gvec, const float* __restrict__ bvec,
                     const float* __restrict__ Wq, half_t* __restrict__ q_h) {
  __shared__ float sl[8][256];
  int b = blockIdx.x, tid = threadIdx.x;
  const float* src = slots_buf + (size_t)b * NQ * DS;
  for (int i = tid; i < NQ * DS; i += 256) sl[i >> 8][i & 255] = src[i];
  __syncthreads();
  int w = __builtin_amdgcn_readfirstlane(tid >> 5);   // wave w owns row w
  int lane = tid & 31;
  {
    float s = 0.f, s2 = 0.f;
#pragma unroll
    for (int i = 0; i < 8; ++i) { float x = sl[w][lane + 32 * i]; s += x; s2 += x * x; }
    for (int m = 1; m < 32; m <<= 1) { s += __shfl_xor(s, m, 32); s2 += __shfl_xor(s2, m, 32); }
    float mean = s * (1.f / 256.f);
    float var  = s2 * (1.f / 256.f) - mean * mean;
    float rstd = rsqrtf(var + LN_EPS);
#pragma unroll
    for (int i = 0; i < 8; ++i) {
      int c = lane + 32 * i;
      sl[w][c] = (sl[w][c] - mean) * rstd * gvec[c] + bvec[c];
    }
  }
  __syncthreads();
  int o = tid;
  float acc[8];
#pragma unroll
  for (int qi = 0; qi < 8; ++qi) acc[qi] = 0.f;
  const float* wr = Wq + (size_t)o * DS;
  for (int c = 0; c < DS; ++c) {
    float wv = wr[c];
#pragma unroll
    for (int qi = 0; qi < 8; ++qi) acc[qi] += sl[qi][c] * wv;
  }
  int h = o >> 6, d = o & 63;
#pragma unroll
  for (int qi = 0; qi < 8; ++qi)
    q_h[((size_t)(b * NHEAD + h) * NQ + qi) * DH + d] = (half_t)acc[qi];
}

// ---------------- k3: attn logits via WMMA ----------------
__global__ void k3_logits(const half_t* __restrict__ k_h, const half_t* __restrict__ q_h,
                          float* __restrict__ logits) {
  int bid = blockIdx.x;                     // ((b*4+h)*32 + kb)
  int b = bid >> 7, h = (bid >> 5) & 3, kb = bid & 31;
  int w = __builtin_amdgcn_readfirstlane(threadIdx.x >> 5);
  int lane = threadIdx.x & 31;
  int rowbase = kb * 128 + w * 16;
  const half_t* abase = k_h + ((size_t)b * NKV + rowbase) * DS + h * DH;
  const half_t* qbase = q_h + (size_t)(b * NHEAD + h) * NQ * DH;
  v8f acc = {};
#pragma unroll
  for (int kc = 0; kc < 2; ++kc) {
    v16h af = load_a_frag(abase + kc * 32, DS, lane);
    v16h bf = load_bt_frag(qbase + kc * 32, DH, lane);
    acc = wmma_f16(af, bf, acc);
  }
  int n = lane & 15, h8 = lane >> 4;
  if (n < NQ) {
    float* lg = logits + (size_t)(b * NHEAD + h) * NKV * NQ;
#pragma unroll
    for (int rr = 0; rr < 8; ++rr)
      lg[(size_t)(rowbase + h8 * 8 + rr) * NQ + n] = acc[rr];
  }
}

// ---------------- k4: joint softmax over (h,q), attn_vis, partial column sums ----------------
__global__ void k4_softmax(const float* __restrict__ logits, float* __restrict__ vis,
                           float* __restrict__ mxbuf, float* __restrict__ denbuf,
                           float* __restrict__ P) {
  __shared__ float wpart[8][32];
  int b = blockIdx.x >> 4, kb = blockIdx.x & 15;
  int tid = threadIdx.x;
  int k = kb * 256 + tid;
  float l[32];
  float mx = -1e30f;
#pragma unroll
  for (int h = 0; h < 4; ++h) {
    const float* p = logits + ((size_t)(b * 4 + h) * NKV + k) * NQ;
#pragma unroll
    for (int q = 0; q < 8; ++q) { float x = p[q]; l[h * 8 + q] = x; mx = fmaxf(mx, x); }
  }
  float den = 0.f;
#pragma unroll
  for (int i = 0; i < 32; ++i) { float e = expf(l[i] - mx); l[i] = e; den += e; }
  mxbuf[(size_t)b * NKV + k] = mx;
  denbuf[(size_t)b * NKV + k] = den;
  float inv = 1.f / den;
#pragma unroll
  for (int q = 0; q < 8; ++q) {
    float vsum = l[q] + l[8 + q] + l[16 + q] + l[24 + q];
    vis[((size_t)b * NKV + k) * NQ + q] = vsum * inv;
  }
  int w = tid >> 5, lane = tid & 31;
  for (int i = 0; i < 32; ++i) {
    float v = l[i] * inv + EPS_A;
    for (int m = 16; m >= 1; m >>= 1) v += __shfl_xor(v, m, 32);
    if (lane == 0) wpart[w][i] = v;
  }
  __syncthreads();
  if (tid < 32) {
    float s = 0.f;
#pragma unroll
    for (int ww = 0; ww < 8; ++ww) s += wpart[ww][tid];
    P[((size_t)b * 16 + kb) * 32 + tid] = s;
  }
}

__global__ void k4b_sreduce(const float* __restrict__ P, float* __restrict__ S) {
  int b = blockIdx.x, t = threadIdx.x;     // 32 threads
  float s = 0.f;
#pragma unroll
  for (int kb = 0; kb < 16; ++kb) s += P[((size_t)b * 16 + kb) * 32 + t];
  S[b * 32 + t] = s;
}

// ---------------- k5: renormalize over kv, store transposed fp16 a^T ----------------
__global__ void k5_norm(const float* __restrict__ logits, const float* __restrict__ mxbuf,
                        const float* __restrict__ denbuf, const float* __restrict__ S,
                        half_t* __restrict__ a_t) {
  int idx = blockIdx.x * 256 + threadIdx.x;   // B*H*NKV = 524288
  int b = idx >> 14;
  int h = (idx >> 12) & 3;
  int k = idx & 4095;
  float mx  = mxbuf[(size_t)b * NKV + k];
  float inv = 1.f / denbuf[(size_t)b * NKV + k];
  const float* p = logits + ((size_t)(b * 4 + h) * NKV + k) * NQ;
#pragma unroll
  for (int q = 0; q < 8; ++q) {
    float a = (expf(p[q] - mx) * inv + EPS_A) / S[b * 32 + h * 8 + q];
    a_t[((size_t)((b * 4 + h) * NQ + q)) * NKV + k] = (half_t)a;
  }
}

// ---------------- k6: updates = a^T x v via WMMA ----------------
__global__ void k6_updates(const half_t* __restrict__ a_t, const half_t* __restrict__ v_h,
                           float* __restrict__ U) {
  int b = blockIdx.x >> 2, h = blockIdx.x & 3;
  int ct = __builtin_amdgcn_readfirstlane(threadIdx.x >> 5);  // 4 waves, one col tile each
  int lane = threadIdx.x & 31;
  int n = lane & 15, h16 = lane >> 4;
  const half_t* abase = a_t + (size_t)(b * 4 + h) * NQ * NKV;
  const half_t* vb0 = v_h + (size_t)b * NKV * DS + h * DH + ct * 16 + n;
  v8f acc = {};
#pragma unroll 1
  for (int kc = 0; kc < 128; ++kc) {
    if (kc + 4 < 128)  // wave-uniform; prefetch v 4 k-chunks ahead (global_prefetch_b8)
      __builtin_prefetch(vb0 + (size_t)((kc + 4) * 32 + h16 * 16) * DS, 0, 1);
    v16h af = load_a_frag(abase + kc * 32, NKV, lane);
    v16h bf;
    const half_t* vb = vb0 + (size_t)(kc * 32 + h16 * 16) * DS;
#pragma unroll
    for (int e = 0; e < 16; ++e) bf[e] = vb[(size_t)e * DS];
    acc = wmma_f16(af, bf, acc);
  }
  if (h16 == 0) {  // rows 0..7 are the valid queries
#pragma unroll
    for (int rr = 0; rr < 8; ++rr)
      U[((size_t)b * NQ + rr) * DS + h * DH + ct * 16 + n] = acc[rr];
  }
}

// ---------------- k7: GRU cell + LN + exact-GELU MLP residual ----------------
__global__ void k7_gru_mlp(const float* __restrict__ U, const float* __restrict__ Wih,
                           const float* __restrict__ Whh, const float* __restrict__ bih,
                           const float* __restrict__ bhh,
                           const float* __restrict__ g_mlp, const float* __restrict__ b_mlp,
                           const float* __restrict__ W1, const float* __restrict__ b1,
                           const float* __restrict__ W2, const float* __restrict__ b2,
                           float* __restrict__ slots_buf, float* __restrict__ out_slots) {
  __shared__ float u[8][256], hp[8][256], ns[8][256];
  __shared__ float hmid[8][1024];
  int b = blockIdx.x, tid = threadIdx.x;
  const float* ub = U + (size_t)b * NQ * DS;
  const float* hb = slots_buf + (size_t)b * NQ * DS;
  for (int i = tid; i < NQ * DS; i += 256) {
    u[i >> 8][i & 255] = ub[i];
    hp[i >> 8][i & 255] = hb[i];
  }
  __syncthreads();
  int c = tid;
  const float* wr = Wih + (size_t)c * DS;
  const float* wz = Wih + (size_t)(DS + c) * DS;
  const float* wn = Wih + (size_t)(2 * DS + c) * DS;
  const float* vr = Whh + (size_t)c * DS;
  const float* vz = Whh + (size_t)(DS + c) * DS;
  const float* vn = Whh + (size_t)(2 * DS + c) * DS;
#pragma unroll 1
  for (int qi = 0; qi < 8; ++qi) {
    float xr = bih[c], xz = bih[DS + c], xn = bih[2 * DS + c];
    float hr = bhh[c], hz = bhh[DS + c], hn = bhh[2 * DS + c];
    for (int i = 0; i < DS; ++i) {
      float uu = u[qi][i], hh = hp[qi][i];
      xr += uu * wr[i]; xz += uu * wz[i]; xn += uu * wn[i];
      hr += hh * vr[i]; hz += hh * vz[i]; hn += hh * vn[i];
    }
    float rg = 1.f / (1.f + expf(-(xr + hr)));
    float zg = 1.f / (1.f + expf(-(xz + hz)));
    float ng = tanhf(xn + rg * hn);
    ns[qi][c] = (1.f - zg) * ng + zg * hp[qi][c];
  }
  __syncthreads();
  // LN(ln_mlp) into u (u is dead now)
  int w = __builtin_amdgcn_readfirstlane(tid >> 5);
  int lane = tid & 31;
  {
    float s = 0.f, s2 = 0.f;
#pragma unroll
    for (int i = 0; i < 8; ++i) { float x = ns[w][lane + 32 * i]; s += x; s2 += x * x; }
    for (int m = 1; m < 32; m <<= 1) { s += __shfl_xor(s, m, 32); s2 += __shfl_xor(s2, m, 32); }
    float mean = s * (1.f / 256.f);
    float var  = s2 * (1.f / 256.f) - mean * mean;
    float rstd = rsqrtf(var + LN_EPS);
#pragma unroll
    for (int i = 0; i < 8; ++i) {
      int cc = lane + 32 * i;
      u[w][cc] = (ns[w][cc] - mean) * rstd * g_mlp[cc] + b_mlp[cc];
    }
  }
  __syncthreads();
  // MLP layer 1: [8,1024], exact GELU
#pragma unroll 1
  for (int i = 0; i < 32; ++i) {
    int oi = tid + i * 256;
    int row = oi >> 10, col = oi & 1023;
    const float* wp = W1 + (size_t)col * DS;
    float acc = b1[col];
    for (int cc2 = 0; cc2 < DS; ++cc2) acc += u[row][cc2] * wp[cc2];
    hmid[row][col] = 0.5f * acc * (1.f + erff(acc * 0.70710678118654752f));
  }
  __syncthreads();
  // MLP layer 2 + residual
  const float* w2r = W2 + (size_t)tid * DMLP;
#pragma unroll 1
  for (int qi = 0; qi < 8; ++qi) {
    float acc = b2[tid];
    for (int jj = 0; jj < DMLP; ++jj) acc += hmid[qi][jj] * w2r[jj];
    float res = ns[qi][tid] + acc;
    slots_buf[(size_t)b * NQ * DS + qi * DS + tid] = res;
    out_slots[(size_t)b * NQ * DS + qi * DS + tid] = res;
  }
}

extern "C" void kernel_launch(void* const* d_in, const int* in_sizes, int n_in,
                              void* d_out, int out_size, void* d_ws, size_t ws_size,
                              hipStream_t stream) {
  (void)in_sizes; (void)n_in; (void)out_size; (void)ws_size;
  const float* inputs     = (const float*)d_in[0];
  const float* slots_init = (const float*)d_in[1];
  const float* ln_in_g    = (const float*)d_in[2];
  const float* ln_in_b    = (const float*)d_in[3];
  const float* ln_slot_g  = (const float*)d_in[4];
  const float* ln_slot_b  = (const float*)d_in[5];
  const float* ln_mlp_g   = (const float*)d_in[6];
  const float* ln_mlp_b   = (const float*)d_in[7];
  const float* Wq   = (const float*)d_in[8];
  const float* Wk   = (const float*)d_in[9];
  const float* Wv   = (const float*)d_in[10];
  const float* gWih = (const float*)d_in[11];
  const float* gWhh = (const float*)d_in[12];
  const float* gbih = (const float*)d_in[13];
  const float* gbhh = (const float*)d_in[14];
  const float* W1   = (const float*)d_in[15];
  const float* b1   = (const float*)d_in[16];
  const float* W2   = (const float*)d_in[17];
  const float* b2   = (const float*)d_in[18];

  float* out        = (float*)d_out;
  float* out_slots  = out;
  float* out_vis    = out + 65536;
  float* out_logits = out + 65536 + 1048576;

  char* ws = (char*)d_ws;
  size_t o = 0;
  auto alloc = [&](size_t bytes) {
    size_t r = o; o = (o + bytes + 255) & ~(size_t)255; return r;
  };
  half_t* wk_h  = (half_t*)(ws + alloc(65536 * 2));
  half_t* wv_h  = (half_t*)(ws + alloc(65536 * 2));
  half_t* q_h   = (half_t*)(ws + alloc((65536 + 512) * 2));                 // + pad rows
  half_t* k_h   = (half_t*)(ws + alloc((size_t)B_SZ * NKV * DS * 2));       // 64 MB
  half_t* v_h   = (half_t*)(ws + alloc((size_t)B_SZ * NKV * DS * 2));       // 64 MB
  half_t* a_t   = (half_t*)(ws + alloc(((size_t)B_SZ * NHEAD * NQ * NKV + NQ * NKV) * 2));
  float*  mxbuf = (float*)(ws + alloc((size_t)B_SZ * NKV * 4));
  float*  denbuf= (float*)(ws + alloc((size_t)B_SZ * NKV * 4));
  float*  P     = (float*)(ws + alloc((size_t)B_SZ * 16 * 32 * 4));
  float*  S     = (float*)(ws + alloc(1024 * 4));
  float*  U     = (float*)(ws + alloc(65536 * 4));
  float*  slots_buf = (float*)(ws + alloc(65536 * 4));

  k0_prep<<<256, 256, 0, stream>>>(Wk, Wv, slots_init, wk_h, wv_h, q_h, a_t, slots_buf);
  k1_ln_kv<<<(B_SZ * NKV) / 32, 256, 0, stream>>>(inputs, ln_in_g, ln_in_b,
                                                  wk_h, wv_h, k_h, v_h);
  for (int it = 0; it < N_IT; ++it) {
    k2_q<<<B_SZ, 256, 0, stream>>>(slots_buf, ln_slot_g, ln_slot_b, Wq, q_h);
    k3_logits<<<B_SZ * NHEAD * 32, 256, 0, stream>>>(k_h, q_h, out_logits);
    k4_softmax<<<B_SZ * 16, 256, 0, stream>>>(out_logits, out_vis, mxbuf, denbuf, P);
    k4b_sreduce<<<B_SZ, 32, 0, stream>>>(P, S);
    k5_norm<<<2048, 256, 0, stream>>>(out_logits, mxbuf, denbuf, S, a_t);
    k6_updates<<<B_SZ * NHEAD, 128, 0, stream>>>(a_t, v_h, U);
    k7_gru_mlp<<<B_SZ, 256, 0, stream>>>(U, gWih, gWhh, gbih, gbhh, ln_mlp_g, ln_mlp_b,
                                         W1, b1, W2, b2, slots_buf, out_slots);
  }
}
